// Attention_14448269984385
// MI455X (gfx1250) — matrix operand order repaired and checked by reading, not compiled, not run on hardware
//
#include <hip/hip_runtime.h>
#include <hip/hip_bf16.h>
#include <cstdint>
#include <cstddef>

// ---------------------------------------------------------------------------
// Causal attention w/ RMSNorm + null-KV for MI455X (gfx1250, wave32, WMMA).
// GEMM A-panels are staged into LDS by the Tensor Data Mover (TDM) and the
// math runs through v_wmma_f32_16x16x32_bf16 (f32 accumulate).
// ---------------------------------------------------------------------------

typedef __bf16 v8bf  __attribute__((ext_vector_type(8)));
typedef __bf16 v16bf __attribute__((ext_vector_type(16)));
typedef float  v8f   __attribute__((ext_vector_type(8)));
typedef unsigned int u32x4 __attribute__((ext_vector_type(4)));
typedef int          i32x4 __attribute__((ext_vector_type(4)));
typedef int          i32x8 __attribute__((ext_vector_type(8)));

#define DIMK   1024
#define NTOK   2048
#define BATCH  4
#define NHEADS 8
#define DHEAD  64
#define INNERD 512
#define ROWS   (BATCH * NTOK)   // 8192

// ---- bf16 helpers (RNE, branch-free) --------------------------------------
static __device__ __forceinline__ unsigned short f2bf_bits(float f) {
  unsigned u = __builtin_bit_cast(unsigned, f);
  u += 0x7FFFu + ((u >> 16) & 1u);
  return (unsigned short)(u >> 16);
}
static __device__ __forceinline__ float bf2f(__bf16 h) {
  unsigned u = ((unsigned)__builtin_bit_cast(unsigned short, h)) << 16;
  return __builtin_bit_cast(float, u);
}

// ---- WMMA fragment loads ---------------------------------------------------
// A fragment 16(M) x 32(K) bf16 from row-major src (stride ld elements).
// Lane m = lane&15 ; half = lane>>4.
// elems 0..15 -> K = k0 + half*16 + e
static __device__ __forceinline__ v16bf load_a(const __bf16* base, int ld,
                                               int row0, int k0) {
  int lane = threadIdx.x & 31;
  int m    = lane & 15;
  int kh   = lane >> 4;
  const __bf16* p = base + (size_t)(row0 + m) * ld + k0 + kh * 16;
  v8bf lo = *(const v8bf*)p;
  v8bf hi = *(const v8bf*)(p + 8);
  v16bf r;
#pragma unroll
  for (int i = 0; i < 8; ++i) { r[i] = lo[i]; r[i + 8] = hi[i]; }
  return r;
}

// B fragment 32(K) x 16(N) bf16 where src is "N-major": row = output column,
// stride ld over K. One 32B load per lane.
static __device__ __forceinline__ v16bf load_b(const __bf16* baseT, int ld,
                                               int n0, int k0) {
  int lane = threadIdx.x & 31;
  int nn   = lane & 15;
  int kh   = lane >> 4;
  return *(const v16bf*)(baseT + (size_t)(n0 + nn) * ld + k0 + kh * 16);
}

static __device__ __forceinline__ v8f wmma_bf16(v16bf a, v16bf b, v8f c) {
  return __builtin_amdgcn_wmma_f32_16x16x32_bf16(false, a, false, b,
                                                 (short)0, c, false, false);
}

// ---- Tensor Data Mover: 2D bf16 tile (rows x cols, row stride = cols) -----
// D# layout per CDNA5 ISA.
// This toolchain exposes the 6-arg builtin form (extra zero descriptor group).
static __device__ __forceinline__ void tdm_load_2d_bf16(const void* gsrc,
                                                        unsigned lds_off,
                                                        int rows, int cols) {
  unsigned long long ga = (unsigned long long)(uintptr_t)gsrc;
  u32x4 g0;
  g0[0] = 1u;                                        // count=1, user mode
  g0[1] = lds_off;                                   // lds_addr (bytes)
  g0[2] = (unsigned)(ga & 0xFFFFFFFFu);              // global_addr[31:0]
  g0[3] = (unsigned)((ga >> 32) & 0x01FFFFFFu) | 0x80000000u;  // [56:32]|type=2
  i32x8 g1;
  g1[0] = (int)(1u << 16);                           // data_size = 2 bytes
  g1[1] = (int)(((unsigned)cols & 0xFFFFu) << 16);   // tensor_dim0[15:0]
  g1[2] = (int)((((unsigned)cols >> 16) & 0xFFFFu) | // tensor_dim0[31:16]
                (((unsigned)rows & 0xFFFFu) << 16)); // tensor_dim1[15:0]
  g1[3] = (int)(((unsigned)cols & 0xFFFFu) << 16);   // tile_dim0 (dim1 hi=0)
  g1[4] = rows;                                      // tile_dim1
  g1[5] = cols;                                      // tensor_dim0_stride lo
  g1[6] = 0;
  g1[7] = 0;
  i32x4 z4 = {0, 0, 0, 0};
  i32x8 z8 = {0, 0, 0, 0, 0, 0, 0, 0};
  __builtin_amdgcn_tensor_load_to_lds(g0, g1, z4, z4, z8, 0);
}

// ---------------------------------------------------------------------------
// Kernel 1a/1b: convert + transpose weights to bf16 "N-major".
// ---------------------------------------------------------------------------
__global__ void prep_wqkv_kernel(const float* __restrict__ w,
                                 unsigned short* __restrict__ wT) {
  int idx = blockIdx.x * 256 + threadIdx.x;          // over 1536*1024
  int j = idx / DIMK, k = idx % DIMK;                // wT[j][k] = w[k][j]
  wT[idx] = f2bf_bits(w[(size_t)k * 1536 + j]);
}

__global__ void prep_wout_kernel(const float* __restrict__ w,
                                 unsigned short* __restrict__ wT) {
  int idx = blockIdx.x * 256 + threadIdx.x;          // over 1024*512
  int j = idx / INNERD, k = idx % INNERD;            // wT[j][k] = w[k][j]
  wT[idx] = f2bf_bits(w[(size_t)k * DIMK + j]);
}

// ---------------------------------------------------------------------------
// Kernel 2: RMSNorm (sum-based):  xn = x * rsqrt(sum(x^2)+eps) * gamma * 32
// ---------------------------------------------------------------------------
__global__ __launch_bounds__(256) void rmsnorm_kernel(
    const float* __restrict__ x, const float* __restrict__ gamma,
    unsigned short* __restrict__ xn) {
  int row  = blockIdx.x;
  int tid  = threadIdx.x;
  int lane = tid & 31, wv = tid >> 5;
  const float* xr = x + (size_t)row * DIMK;
  float s = 0.f;
#pragma unroll
  for (int c = tid; c < DIMK; c += 256) { float v = xr[c]; s += v * v; }
#pragma unroll
  for (int off = 16; off >= 1; off >>= 1) s += __shfl_xor(s, off, 32);
  __shared__ float ws[8];
  if (lane == 0) ws[wv] = s;
  __syncthreads();
  float tot = 0.f;
#pragma unroll
  for (int i = 0; i < 8; ++i) tot += ws[i];
  float sc = rsqrtf(tot + 1e-5f) * 32.0f;            // sqrt(1024) = 32
  unsigned short* o = xn + (size_t)row * DIMK;
#pragma unroll
  for (int c = tid; c < DIMK; c += 256) o[c] = f2bf_bits(xr[c] * sc * gamma[c]);
}

// ---------------------------------------------------------------------------
// Kernel 3: QKV GEMM [8192x1024]@[1024x1536] -> scatter q/k/vT (bf16).
// Block = 8 waves sharing one TDM-staged A panel (16 rows x 1024) in LDS;
// each wave owns a 16x64 output tile (n0 = cg*512 + wave*64).
// ---------------------------------------------------------------------------
__global__ __launch_bounds__(256) void gemm_qkv_kernel(
    const __bf16* __restrict__ xn, const __bf16* __restrict__ wT,
    unsigned short* __restrict__ qbuf, unsigned short* __restrict__ kbuf,
    unsigned short* __restrict__ vtbuf) {
  __shared__ __attribute__((aligned(16))) __bf16 tileA[16 * DIMK];  // 32 KB
  int lane = threadIdx.x & 31, wv = threadIdx.x >> 5;
  int mt = blockIdx.x / 3, cg = blockIdx.x % 3;
  int i0 = mt * 16;
  int n0 = cg * 512 + wv * 64;
  int nn = lane & 15, hi = lane >> 4;

  if (wv == 0) {
    tdm_load_2d_bf16(xn + (size_t)i0 * DIMK,
                     (unsigned)(uintptr_t)&tileA[0], 16, DIMK);
    __builtin_amdgcn_s_wait_tensorcnt(0);
  }
  __syncthreads();

  v8f acc[4] = {};
  for (int kk = 0; kk < DIMK; kk += 32) {
    v16bf a = load_a(&tileA[0], DIMK, 0, kk);        // ds_load from LDS
#pragma unroll
    for (int c = 0; c < 4; ++c) {
      v16bf b = load_b(wT, DIMK, n0 + c * 16, kk);
      acc[c] = wmma_bf16(a, b, acc[c]);
    }
  }
#pragma unroll
  for (int c = 0; c < 4; ++c) {
#pragma unroll
    for (int r = 0; r < 8; ++r) {
      int row = i0 + r + 8 * hi;
      int bi = row >> 11, t = row & 2047;
      int j = n0 + c * 16 + nn;
      float v = acc[c][r];
      if (j < 512) {            // Q (pre-scaled by dhead^-0.5 = 1/8)
        int h = j >> 6, d = j & 63;
        qbuf[(((size_t)(bi * NHEADS + h) * NTOK) + t) * DHEAD + d] =
            f2bf_bits(v * 0.125f);
      } else if (j < 1024) {    // K
        int jj = j - 512; int h = jj >> 6, d = jj & 63;
        kbuf[(((size_t)(bi * NHEADS + h) * NTOK) + t) * DHEAD + d] =
            f2bf_bits(v);
      } else {                  // V, stored transposed [b,h,64,n]
        int jj = j - 1024; int h = jj >> 6, d = jj & 63;
        vtbuf[((size_t)(bi * NHEADS + h) * DHEAD + d) * NTOK + t] =
            f2bf_bits(v);
      }
    }
  }
}

// ---------------------------------------------------------------------------
// Kernel 4: causal flash attention with null-KV token.
// One wave per 16-query tile; key tiles of 32; online softmax. Waves are
// autonomous (different causal trip counts) -> no block barriers here.
// ---------------------------------------------------------------------------
__global__ __launch_bounds__(256) void attn_kernel(
    const __bf16* __restrict__ qbuf, const __bf16* __restrict__ kbuf,
    const __bf16* __restrict__ vtbuf, const float* __restrict__ nullkv,
    unsigned short* __restrict__ ctx) {
  __shared__ unsigned short pst[8][16][32];          // per-wave P staging
  int lane = threadIdx.x & 31;
  int wv   = threadIdx.x >> 5;
  int wgid = blockIdx.x * 8 + wv;
  int tile = wgid & 127;                             // 2048/16 query tiles
  int bh   = wgid >> 7;                              // 0..31
  int h    = bh & 7;
  int i0   = tile * 16;
  const __bf16* qb  = qbuf + (size_t)bh * NTOK * DHEAD;
  const __bf16* kb  = kbuf + (size_t)bh * NTOK * DHEAD;
  const __bf16* vtb = vtbuf + (size_t)bh * DHEAD * NTOK;
  const float* nk = nullkv + h * 2 * DHEAD;          // null key
  const float* nv = nk + DHEAD;                      // null value
  int nn = lane & 15, hi = lane >> 4;

  v16bf aq0 = load_a(qb, DHEAD, i0, 0);
  v16bf aq1 = load_a(qb, DHEAD, i0, 32);

  // null-token score for row (lane&15): q already carries the 1/8 scale.
  float snull = 0.f;
  {
    const __bf16* qr = qb + (size_t)(i0 + nn) * DHEAD;
#pragma unroll
    for (int d = 0; d < DHEAD; ++d) snull += bf2f(qr[d]) * nk[d];
  }

  float mrun[8], lrun[8];
  v8f o0 = {}, o1 = {}, o2 = {}, o3 = {};
  float nv0 = nv[nn], nv1 = nv[16 + nn], nv2 = nv[32 + nn], nv3 = nv[48 + nn];
#pragma unroll
  for (int r = 0; r < 8; ++r) {
    mrun[r] = __shfl(snull, r + 8 * hi, 32);         // init m = s_null
    lrun[r] = 1.0f;                                  // exp(s_null - m) = 1
    o0[r] = nv0; o1[r] = nv1; o2[r] = nv2; o3[r] = nv3;  // o = 1 * null_v
  }

  int jend = i0 + 16;                                // causal key count
  for (int j0 = 0; j0 < jend; j0 += 32) {
    v16bf bk00 = load_b(kb, DHEAD, j0,      0);
    v16bf bk01 = load_b(kb, DHEAD, j0,      32);
    v16bf bk10 = load_b(kb, DHEAD, j0 + 16, 0);
    v16bf bk11 = load_b(kb, DHEAD, j0 + 16, 32);
    v8f s0 = {}, s1 = {};
    s0 = wmma_bf16(aq0, bk00, s0);
    s0 = wmma_bf16(aq1, bk01, s0);
    s1 = wmma_bf16(aq0, bk10, s1);
    s1 = wmma_bf16(aq1, bk11, s1);

#pragma unroll
    for (int r = 0; r < 8; ++r) {
      int rowg = i0 + r + 8 * hi;
      int kg   = j0 + nn;
      float v0 = (kg      > rowg) ? -3.0e38f : s0[r];   // causal mask
      float v1 = (kg + 16 > rowg) ? -3.0e38f : s1[r];
      float t = fmaxf(v0, v1);
#pragma unroll
      for (int off = 8; off >= 1; off >>= 1) t = fmaxf(t, __shfl_xor(t, off, 32));
      float newm = fmaxf(mrun[r], t);
      float fac  = __expf(mrun[r] - newm);
      float p0   = __expf(v0 - newm);
      float p1   = __expf(v1 - newm);
      float ps   = p0 + p1;
#pragma unroll
      for (int off = 8; off >= 1; off >>= 1) ps += __shfl_xor(ps, off, 32);
      lrun[r] = lrun[r] * fac + ps;
      mrun[r] = newm;
      o0[r] *= fac; o1[r] *= fac; o2[r] *= fac; o3[r] *= fac;
      pst[wv][r + 8 * hi][nn]      = f2bf_bits(p0);  // stage P (C->A relayout)
      pst[wv][r + 8 * hi][16 + nn] = f2bf_bits(p1);
    }
    asm volatile("s_wait_dscnt 0" ::: "memory");     // wave-local LDS drain

    // read P back as an A fragment (16 x 32)
    const unsigned short* prow = &pst[wv][nn][0];
    v8bf plo = *(const v8bf*)(prow + hi * 16);
    v8bf phi = *(const v8bf*)(prow + hi * 16 + 8);
    v16bf ap;
#pragma unroll
    for (int i = 0; i < 8; ++i) { ap[i] = plo[i]; ap[i + 8] = phi[i]; }

    v16bf bv0 = load_b(vtb, NTOK, 0,  j0);           // V^T: rows = head dim
    v16bf bv1 = load_b(vtb, NTOK, 16, j0);
    v16bf bv2 = load_b(vtb, NTOK, 32, j0);
    v16bf bv3 = load_b(vtb, NTOK, 48, j0);
    o0 = wmma_bf16(ap, bv0, o0);
    o1 = wmma_bf16(ap, bv1, o1);
    o2 = wmma_bf16(ap, bv2, o2);
    o3 = wmma_bf16(ap, bv3, o3);
  }

#pragma unroll
  for (int r = 0; r < 8; ++r) {
    float inv = 1.0f / lrun[r];
    int rowg = i0 + r + 8 * hi;
    size_t crow = (size_t)((bh >> 3) * NTOK + rowg) * INNERD + h * DHEAD + nn;
    ctx[crow +  0] = f2bf_bits(o0[r] * inv);
    ctx[crow + 16] = f2bf_bits(o1[r] * inv);
    ctx[crow + 32] = f2bf_bits(o2[r] * inv);
    ctx[crow + 48] = f2bf_bits(o3[r] * inv);
  }
}

// ---------------------------------------------------------------------------
// Kernel 5: out-proj GEMM [8192x512]@[512x1024] -> f32 d_out.
// Same TDM-staged A panel structure (16 x 512 in LDS, 8 waves share).
// ---------------------------------------------------------------------------
__global__ __launch_bounds__(256) void gemm_out_kernel(
    const __bf16* __restrict__ ctx, const __bf16* __restrict__ wT,
    float* __restrict__ out) {
  __shared__ __attribute__((aligned(16))) __bf16 tileA[16 * INNERD];  // 16 KB
  int lane = threadIdx.x & 31, wv = threadIdx.x >> 5;
  int mt = blockIdx.x >> 1, cg = blockIdx.x & 1;
  int i0 = mt * 16;
  int n0 = cg * 512 + wv * 64;
  int nn = lane & 15, hi = lane >> 4;

  if (wv == 0) {
    tdm_load_2d_bf16(ctx + (size_t)i0 * INNERD,
                     (unsigned)(uintptr_t)&tileA[0], 16, INNERD);
    __builtin_amdgcn_s_wait_tensorcnt(0);
  }
  __syncthreads();

  v8f acc[4] = {};
  for (int kk = 0; kk < INNERD; kk += 32) {
    v16bf a = load_a(&tileA[0], INNERD, 0, kk);
#pragma unroll
    for (int c = 0; c < 4; ++c) {
      v16bf b = load_b(wT, INNERD, n0 + c * 16, kk);
      acc[c] = wmma_bf16(a, b, acc[c]);
    }
  }
#pragma unroll
  for (int c = 0; c < 4; ++c) {
#pragma unroll
    for (int r = 0; r < 8; ++r) {
      int row = i0 + r + 8 * hi;
      out[(size_t)row * DIMK + n0 + c * 16 + nn] = acc[c][r];
    }
  }
}

// ---------------------------------------------------------------------------
// Host-side launcher.
// Inputs: 0:x 1:mask(all-true, ignored) 2:gamma 3:null_kv 4:w_qkv 5:w_out
// ---------------------------------------------------------------------------
extern "C" void kernel_launch(void* const* d_in, const int* in_sizes, int n_in,
                              void* d_out, int out_size, void* d_ws,
                              size_t ws_size, hipStream_t stream) {
  const float* x      = (const float*)d_in[0];
  const float* gamma  = (const float*)d_in[2];
  const float* nullkv = (const float*)d_in[3];
  const float* w_qkv  = (const float*)d_in[4];
  const float* w_out  = (const float*)d_in[5];
  float* out = (float*)d_out;

  char* ws = (char*)d_ws;
  size_t off = 0;
  __bf16* wqkvT = (__bf16*)(ws + off); off += (size_t)1536 * DIMK   * 2;  // 3 MB
  __bf16* woutT = (__bf16*)(ws + off); off += (size_t)DIMK * INNERD * 2;  // 1 MB
  __bf16* xn    = (__bf16*)(ws + off); off += (size_t)ROWS * DIMK   * 2;  // 16 MB
  __bf16* qbuf  = (__bf16*)(ws + off); off += (size_t)BATCH*NHEADS*NTOK*DHEAD*2;
  __bf16* kbuf  = (__bf16*)(ws + off); off += (size_t)BATCH*NHEADS*NTOK*DHEAD*2;
  __bf16* vtbuf = (__bf16*)(ws + off); off += (size_t)BATCH*NHEADS*NTOK*DHEAD*2;
  __bf16* ctx   = (__bf16*)(ws + off); off += (size_t)ROWS * INNERD * 2;  // 8 MB

  prep_wqkv_kernel<<<(1536 * DIMK) / 256, 256, 0, stream>>>(
      w_qkv, (unsigned short*)wqkvT);
  prep_wout_kernel<<<(DIMK * INNERD) / 256, 256, 0, stream>>>(
      w_out, (unsigned short*)woutT);
  rmsnorm_kernel<<<ROWS, 256, 0, stream>>>(x, gamma, (unsigned short*)xn);
  gemm_qkv_kernel<<<512 * 3, 256, 0, stream>>>(
      xn, wqkvT, (unsigned short*)qbuf, (unsigned short*)kbuf,
      (unsigned short*)vtbuf);
  attn_kernel<<<(BATCH * NHEADS * (NTOK / 16)) / 8, 256, 0, stream>>>(
      qbuf, kbuf, vtbuf, nullkv, (unsigned short*)ctx);
  gemm_out_kernel<<<512 * 2, 256, 0, stream>>>(ctx, woutT, out);
}
